// LSTM_RNN_17824114279180
// MI455X (gfx1250) — matrix operand order repaired and checked by reading, not compiled, not run on hardware
//
#include <hip/hip_runtime.h>
#include <hip/hip_bf16.h>
#include <math.h>

// ---- problem constants ----
#define BB 128      // batch
#define TT 512      // timesteps
#define EE 256      // embedding dim
#define UU 512      // hidden units
#define KK 768      // E + U (concat GEMM K)
#define NN 2048     // 4*U gate columns
#define VOCAB 32000

typedef __attribute__((ext_vector_type(16))) __bf16 v16bf;
typedef __attribute__((ext_vector_type(8)))  float  v8f;

__device__ __forceinline__ unsigned short f2bf(float f) {
    unsigned int u = __float_as_uint(f);
    unsigned int r = ((u >> 16) & 1u) + 0x7FFFu;   // round to nearest even
    return (unsigned short)((u + r) >> 16);
}

// -------------------------------------------------------------------------
// Prep 1: swizzle concat(Wx,Wh) fp32 -> bf16 directly into WMMA B-fragment
// order:  Wswz[bx][s][g][lane][e], bx=unit-group(32), s=K-slice(24),
// g=gate(4), lane(32), e(16).  Lane l holds col g*512+bx*16+(l&15),
// K = s*32 + (e>>3)*16 + (l>>4)*8 + (e&7).  Also zero h0 (bf16) and c0.
// -------------------------------------------------------------------------
__global__ __launch_bounds__(256) void prep_w_kernel(
    const float* __restrict__ Wx, const float* __restrict__ Wh,
    unsigned short* __restrict__ Wswz,
    unsigned short* __restrict__ hbf0, float* __restrict__ cst)
{
    const int idx = blockIdx.x * 256 + threadIdx.x;     // 0 .. 768*2048-1
    if (idx < KK * NN) {
        const int e  = idx & 15;
        const int q  = idx >> 4;
        const int l  = q & 31;
        const int q2 = q >> 5;
        const int g  = q2 & 3;
        const int q3 = q2 >> 2;
        const int s  = q3 % 24;
        const int bx = q3 / 24;
        const int K    = (s << 5) + ((e >> 3) << 4) + ((l >> 4) << 3) + (e & 7);
        const int gcol = (g << 9) + (bx << 4) + (l & 15);
        const float v = (K < EE) ? Wx[K * NN + gcol] : Wh[(K - EE) * NN + gcol];
        Wswz[idx] = f2bf(v);
    }
    if (idx < BB * UU) { hbf0[idx] = 0; cst[idx] = 0.0f; }
}

// Prep 2: embedding table fp32 -> bf16 (row-major unchanged), once.
__global__ __launch_bounds__(256) void prep_emb_kernel(
    const float* __restrict__ emb, unsigned short* __restrict__ embbf, int n)
{
    for (int i = blockIdx.x * 256 + threadIdx.x; i < n; i += gridDim.x * 256)
        embbf[i] = f2bf(emb[i]);
}

// -------------------------------------------------------------------------
// One LSTM timestep.  Grid: 32 blocks (unit groups of 16) x 256 threads.
// Wave w owns rows 16w..16w+15 and units u0..u0+15 for ALL four gates:
// 4 accumulator tiles, K-loop of 24 slices, no LDS, no barriers.
// Fragments are double-buffered in registers: iteration s+1's loads are
// issued before iteration s's WMMAs so a full slice is always in flight.
// -------------------------------------------------------------------------
__global__ __launch_bounds__(256) void lstm_step_kernel(
    const int*            __restrict__ sent,
    const unsigned short* __restrict__ embbf,
    const unsigned short* __restrict__ Wswz,
    const float*          __restrict__ bias,
    const unsigned short* __restrict__ hbf_in,
    unsigned short*       __restrict__ hbf_out,
    float*                __restrict__ hf,
    float*                __restrict__ cst,
    int t)
{
    const int tid  = threadIdx.x;
    const int lane = tid & 31;
    const int wv   = tid >> 5;                  // 0..7 -> rows 16*wv..+15
    const int u0   = blockIdx.x << 4;           // 16 units per block

    const int rowA = (wv << 4) + (lane & 15);
    const int offA = (lane >> 4) << 3;          // lanes 16-31: K +8 within half
    const int tok  = sent[rowA * TT + t];

    const v8f vzero = {0.f,0.f,0.f,0.f,0.f,0.f,0.f,0.f};
    v8f acc[4] = {vzero, vzero, vzero, vzero};

    const unsigned short* __restrict__ wp =
        Wswz + (size_t)blockIdx.x * (24 * 4 * 32 * 16) + lane * 16;
    const unsigned short* __restrict__ ax = embbf + (size_t)tok  * EE + offA;
    const unsigned short* __restrict__ ah = hbf_in + (size_t)rowA * UU + offA;

    union AF { uint4 u[2]; v16bf v; };
    union BF { uint4 u[2]; v16bf v; };
    AF fa[2];
    BF fb[2][4];

    // prologue: load slice 0
    {
        const unsigned short* ap = ax;                 // s=0 < 8 -> emb source
        fa[0].u[0] = *reinterpret_cast<const uint4*>(ap);
        fa[0].u[1] = *reinterpret_cast<const uint4*>(ap + 16);
        const unsigned short* wps = wp;
        #pragma unroll
        for (int g = 0; g < 4; ++g) {
            fb[0][g].u[0] = *reinterpret_cast<const uint4*>(wps + g * 512);
            fb[0][g].u[1] = *reinterpret_cast<const uint4*>(wps + g * 512 + 8);
        }
    }

    #pragma unroll
    for (int s = 0; s < 24; ++s) {
        const int cur = s & 1;
        const int nxt = cur ^ 1;
        if (s < 23) {                                  // prefetch slice s+1
            const int sn = s + 1;
            const unsigned short* ap =
                (sn < 8) ? (ax + (sn << 5)) : (ah + ((sn - 8) << 5));
            fa[nxt].u[0] = *reinterpret_cast<const uint4*>(ap);
            fa[nxt].u[1] = *reinterpret_cast<const uint4*>(ap + 16);
            const unsigned short* wps = wp + (size_t)sn * (4 * 32 * 16);
            #pragma unroll
            for (int g = 0; g < 4; ++g) {
                fb[nxt][g].u[0] = *reinterpret_cast<const uint4*>(wps + g * 512);
                fb[nxt][g].u[1] = *reinterpret_cast<const uint4*>(wps + g * 512 + 8);
            }
        }
        #pragma unroll
        for (int g = 0; g < 4; ++g) {
            acc[g] = __builtin_amdgcn_wmma_f32_16x16x32_bf16(
                false, fa[cur].v, false, fb[cur][g].v, (short)0, acc[g],
                false, false);
        }
    }

    // ---- fused LSTM cell update: lane owns (row r, unit) pairs ----
    const int unit = u0 + (lane & 15);
    const float bi = bias[unit];
    const float bf = bias[UU + unit];
    const float bg = bias[2 * UU + unit];
    const float bo = bias[3 * UU + unit];
    const int rbase = (wv << 4) + ((lane >> 4) << 3);
    #pragma unroll
    for (int r = 0; r < 8; ++r) {
        const int row = rbase + r;
        const float zi = acc[0][r] + bi;
        const float zf = acc[1][r] + bf;
        const float zg = acc[2][r] + bg;
        const float zo = acc[3][r] + bo;
        const float ig = 1.f / (1.f + __expf(-zi));
        const float fg = 1.f / (1.f + __expf(-zf));
        const float gg = tanhf(zg);
        const float og = 1.f / (1.f + __expf(-zo));
        const size_t idx = (size_t)row * UU + unit;
        const float cn = fg * cst[idx] + ig * gg;
        cst[idx] = cn;
        const float hn = og * tanhf(cn);
        hf[idx] = hn;
        hbf_out[idx] = f2bf(hn);
    }
}

// -------------------------------------------------------------------------
// MLP head (tiny): three plain fp32 kernels.
// -------------------------------------------------------------------------
__global__ __launch_bounds__(256) void head1_kernel(
    const float* __restrict__ hf, const float* __restrict__ W1,
    const float* __restrict__ b1, float* __restrict__ h1)
{
    const int idx = blockIdx.x * 256 + threadIdx.x;   // 128*128
    const int row = idx >> 7, j = idx & 127;
    float s = b1[j];
    for (int k = 0; k < UU; ++k) s += hf[row * UU + k] * W1[k * 128 + j];
    h1[idx] = fmaxf(s, 0.f);
}

__global__ __launch_bounds__(256) void head2_kernel(
    const float* __restrict__ h1, const float* __restrict__ W2,
    const float* __restrict__ b2, float* __restrict__ h2)
{
    const int idx = blockIdx.x * 256 + threadIdx.x;   // 128*64
    const int row = idx >> 6, j = idx & 63;
    float s = b2[j];
    for (int k = 0; k < 128; ++k) s += h1[row * 128 + k] * W2[k * 64 + j];
    h2[idx] = fmaxf(s, 0.f);
}

__global__ __launch_bounds__(128) void head3_kernel(
    const float* __restrict__ h2, const float* __restrict__ W3,
    const float* __restrict__ b3, float* __restrict__ out)
{
    const int row = threadIdx.x;                      // 128
    float s = b3[0];
    for (int k = 0; k < 64; ++k) s += h2[row * 64 + k] * W3[k];
    out[row] = 1.f / (1.f + __expf(-s));
}

// -------------------------------------------------------------------------
extern "C" void kernel_launch(void* const* d_in, const int* in_sizes, int n_in,
                              void* d_out, int out_size, void* d_ws, size_t ws_size,
                              hipStream_t stream)
{
    const int*   sent = (const int*)  d_in[0];
    const float* emb  = (const float*)d_in[1];
    const float* Wx   = (const float*)d_in[2];
    const float* Wh   = (const float*)d_in[3];
    const float* bias = (const float*)d_in[4];
    const float* W1   = (const float*)d_in[5];
    const float* b1   = (const float*)d_in[6];
    const float* W2   = (const float*)d_in[7];
    const float* b2   = (const float*)d_in[8];
    const float* W3   = (const float*)d_in[9];
    const float* b3   = (const float*)d_in[10];
    float* out = (float*)d_out;

    char* ws = (char*)d_ws;
    unsigned short* Wswz  = (unsigned short*)(ws + 0);         // 3,145,728 B
    unsigned short* embbf = (unsigned short*)(ws + 3145728);   // 16,384,000 B
    unsigned short* hbfA  = (unsigned short*)(ws + 19529728);  // 131,072 B
    unsigned short* hbfB  = (unsigned short*)(ws + 19660800);  // 131,072 B
    float*          hf    = (float*)        (ws + 19791872);   // 262,144 B
    float*          cst   = (float*)        (ws + 20054016);   // 262,144 B
    float*          h1    = (float*)        (ws + 20316160);   // 65,536 B
    float*          h2    = (float*)        (ws + 20381696);   // 32,768 B

    // 1) one-time conversions + state zeroing (amortized over 512 steps)
    prep_w_kernel<<<(KK * NN + 255) / 256, 256, 0, stream>>>(Wx, Wh, Wswz, hbfA, cst);
    prep_emb_kernel<<<4096, 256, 0, stream>>>(emb, embbf, VOCAB * EE);

    // 2) serial recurrence, ping-pong bf16 h buffers
    for (int t = 0; t < TT; ++t) {
        unsigned short* hin  = (t & 1) ? hbfB : hbfA;
        unsigned short* hout = (t & 1) ? hbfA : hbfB;
        lstm_step_kernel<<<UU / 16, 256, 0, stream>>>(
            sent, embbf, Wswz, bias, hin, hout, hf, cst, t);
    }

    // 3) MLP head
    head1_kernel<<<(BB * 128) / 256, 256, 0, stream>>>(hf, W1, b1, h1);
    head2_kernel<<<(BB * 64) / 256, 256, 0, stream>>>(h1, W2, b2, h2);
    head3_kernel<<<1, 128, 0, stream>>>(h2, W3, b3, out);
}
